// AttentionLayer_48928267436533
// MI455X (gfx1250) — matrix operand order repaired and checked by reading, not compiled, not run on hardware
//
#include <hip/hip_runtime.h>

typedef __bf16 bf16_t;
typedef __attribute__((ext_vector_type(16))) __bf16 v16bf;
typedef __attribute__((ext_vector_type(8)))  float  v8f;

union ABFrag { v16bf v; uint4 q[2]; };

#define BATCH 4
#define SEQ   2048
#define DIM   1024   // D_IN == D_OUT == 1024

static __device__ __forceinline__ v8f wmma_bf16(const v16bf& a, const v16bf& b, const v8f& c) {
    return __builtin_amdgcn_wmma_f32_16x16x32_bf16(false, a, false, b, (short)0, c, false, false);
}

// ---------------- Kernel 1: X (f32) -> Xb (bf16), same row-major layout ----------------
__global__ __launch_bounds__(256) void cvt_x_kernel(const float* __restrict__ X,
                                                    bf16_t* __restrict__ Xb) {
    size_t i = ((size_t)blockIdx.x * 256 + threadIdx.x) * 4;
    float4 v = *(const float4*)(X + i);
    union { bf16_t h[4]; uint2 u; } u;
    u.h[0] = (bf16_t)v.x; u.h[1] = (bf16_t)v.y;
    u.h[2] = (bf16_t)v.z; u.h[3] = (bf16_t)v.w;
    *(uint2*)(Xb + i) = u.u;
}

// ---------------- Kernel 2: W[d][e] (f32) -> Wt[e][d] (bf16), 3 matrices ----------------
__global__ __launch_bounds__(256) void cvt_w_kernel(const float* __restrict__ Wq,
                                                    const float* __restrict__ Wk,
                                                    const float* __restrict__ Wv,
                                                    bf16_t* __restrict__ Wt) {
    const int e = blockIdx.x;                  // output feature (row of Wt)
    const int z = blockIdx.y;                  // which matrix
    const float* W = (z == 0) ? Wq : (z == 1) ? Wk : Wv;
    const int d0 = threadIdx.x * 4;
    union { bf16_t h[4]; uint2 u; } u;
#pragma unroll
    for (int jj = 0; jj < 4; ++jj)
        u.h[jj] = (bf16_t)W[(size_t)(d0 + jj) * DIM + e];
    *(uint2*)(Wt + (size_t)z * DIM * DIM + (size_t)e * DIM + d0) = u.u;
}

// ---------------- Kernel 3: QKV GEMM:  Y = Xb @ W + bias  (bf16 WMMA, f32 accum) --------
// grid = (8192/128, 1024/256, 3), block = 256 (8 waves, 2x4 grid of 64x64 wave tiles).
// z: 0->Q, 1->K, 2->V(transposed)
__global__ __launch_bounds__(256) void qkv_gemm_kernel(const bf16_t* __restrict__ Xb,
                                                       const bf16_t* __restrict__ WtBase,
                                                       const float* __restrict__ bq,
                                                       const float* __restrict__ bk,
                                                       const float* __restrict__ bv,
                                                       bf16_t* __restrict__ Qb,
                                                       bf16_t* __restrict__ Kb,
                                                       bf16_t* __restrict__ Vt) {
    const int z    = blockIdx.z;
    const bf16_t* W = WtBase + (size_t)z * DIM * DIM;
    const float* bias = (z == 0) ? bq : (z == 1) ? bk : bv;
    const int row0 = blockIdx.x * 128;
    const int col0 = blockIdx.y * 256;
    const int tid  = threadIdx.x;
    const int wave = tid >> 5, lane = tid & 31;
    const int hf   = lane >> 4, l16 = lane & 15;
    const int wm   = wave >> 2, wn = wave & 3;   // 2x4 wave grid, wave tile 64x64

    __shared__ __align__(16) bf16_t aT[128 * 64];  // [row][k], stride 64 (16 KB)
    __shared__ __align__(16) bf16_t bT[256 * 64];  // [e_local][k], stride 64 (32 KB)

    v8f acc[4][4];
#pragma unroll
    for (int i = 0; i < 4; ++i)
#pragma unroll
        for (int j = 0; j < 4; ++j) acc[i][j] = (v8f){};

    for (int k0 = 0; k0 < DIM; k0 += 64) {
        __syncthreads();
#pragma unroll
        for (int it = 0; it < 4; ++it) {
            int e = (tid + it * 256) * 8;
            int r = e >> 6, c = e & 63;
            *(uint4*)(aT + e) = *(const uint4*)(Xb + (size_t)(row0 + r) * DIM + k0 + c);
        }
#pragma unroll
        for (int it = 0; it < 8; ++it) {
            int e = (tid + it * 256) * 8;
            int r = e >> 6, c = e & 63;
            *(uint4*)(bT + e) = *(const uint4*)(W + (size_t)(col0 + r) * DIM + k0 + c);
        }
        __syncthreads();
#pragma unroll
        for (int kk = 0; kk < 64; kk += 32) {
            ABFrag af[4], bfr[4];
#pragma unroll
            for (int i = 0; i < 4; ++i) {
                const bf16_t* p = aT + (wm * 64 + i * 16 + l16) * 64 + kk + hf * 8;
                af[i].q[0] = *(const uint4*)p;
                af[i].q[1] = *(const uint4*)(p + 16);
            }
#pragma unroll
            for (int j = 0; j < 4; ++j) {
                const bf16_t* p = bT + (wn * 64 + j * 16 + l16) * 64 + kk + hf * 8;
                bfr[j].q[0] = *(const uint4*)p;
                bfr[j].q[1] = *(const uint4*)(p + 16);
            }
#pragma unroll
            for (int i = 0; i < 4; ++i)
#pragma unroll
                for (int j = 0; j < 4; ++j)
                    acc[i][j] = wmma_bf16(af[i].v, bfr[j].v, acc[i][j]);
        }
    }

    // bias + store
#pragma unroll
    for (int i = 0; i < 4; ++i)
#pragma unroll
        for (int j = 0; j < 4; ++j) {
            const int e  = col0 + wn * 64 + j * 16 + l16;       // output feature (col)
            const float bvv = bias[e];
            const int rbase = row0 + wm * 64 + i * 16 + 8 * hf; // first of 8 consecutive rows
            if (z == 2) {
                // V transposed: Vt[b][e][n], n consecutive -> one 16B store
                const int bidx = rbase >> 11, n = rbase & (SEQ - 1);
                union { bf16_t h[8]; uint4 q; } u;
#pragma unroll
                for (int ii = 0; ii < 8; ++ii) u.h[ii] = (bf16_t)(acc[i][j][ii] + bvv);
                *(uint4*)(Vt + (((size_t)(bidx * DIM + e)) << 11) + n) = u.q;
            } else {
                bf16_t* dst = (z == 0) ? Qb : Kb;
#pragma unroll
                for (int ii = 0; ii < 8; ++ii)
                    dst[(size_t)(rbase + ii) * DIM + e] = (bf16_t)(acc[i][j][ii] + bvv);
            }
        }
}

// ---------------- Kernel 4: flash attention ----------------
// grid = (SEQ/32, BATCH), block = 256 (8 waves). Block owns 32 query rows (2 tiles).
// Per 128-key block: wave w computes two 16x16 S tiles for keys [j0+16w, j0+16w+16)
// (K fragment reused across both query tiles) -> LDS, cooperative online softmax,
// then each wave accumulates its 128 output columns (V fragment reused across tiles).
// Q is read directly from global (L2-resident across the 16 key-block iterations).
__global__ __launch_bounds__(256) void attn_kernel(const bf16_t* __restrict__ Qb,
                                                   const bf16_t* __restrict__ Kb,
                                                   const bf16_t* __restrict__ Vt,
                                                   float* __restrict__ out) {
    const int qBase = blockIdx.x * 32;
    const int b     = blockIdx.y;
    const int tid   = threadIdx.x;
    const int wave  = tid >> 5, lane = tid & 31;
    const int hf    = lane >> 4, l16 = lane & 15;

    __shared__ __align__(16) float  sS[32 * 128];   // scores f32 (16 KB)
    __shared__ __align__(16) bf16_t pB[32 * 128];   // probabilities bf16 (8 KB)
    __shared__ float aB[32];                        // per-row rescale factor
    __shared__ float lB[32];                        // per-row final denominator

    v8f acc[2][8];
#pragma unroll
    for (int qi = 0; qi < 2; ++qi)
#pragma unroll
        for (int t = 0; t < 8; ++t) acc[qi][t] = (v8f){};
    float m[2] = {-1e30f, -1e30f};
    float l[2] = {0.f, 0.f};

    const bf16_t* Qrow = Qb + (((size_t)(b * SEQ + qBase)) << 10);
    const bf16_t* Krow = Kb + (size_t)b * SEQ * DIM;
    const bf16_t* Vrow = Vt + (size_t)b * DIM * SEQ;

    for (int j0 = 0; j0 < SEQ; j0 += 128) {
        // ---- S = Q @ K^T, two query tiles against this wave's 16-key subtile ----
        v8f sacc[2] = {(v8f){}, (v8f){}};
        {
            const bf16_t* kptr = Krow + (size_t)(j0 + wave * 16 + l16) * DIM;
#pragma unroll 2
            for (int d0 = 0; d0 < DIM; d0 += 32) {
                ABFrag bfr;
                const bf16_t* bp = kptr + d0 + hf * 8;
                bfr.q[0] = *(const uint4*)bp;
                bfr.q[1] = *(const uint4*)(bp + 16);
#pragma unroll
                for (int qi = 0; qi < 2; ++qi) {
                    ABFrag af;
                    const bf16_t* ap = Qrow + ((size_t)(qi * 16 + l16) << 10) + d0 + hf * 8;
                    af.q[0] = *(const uint4*)ap;
                    af.q[1] = *(const uint4*)(ap + 16);
                    sacc[qi] = wmma_bf16(af.v, bfr.v, sacc[qi]);
                }
            }
        }
#pragma unroll
        for (int qi = 0; qi < 2; ++qi)
#pragma unroll
            for (int ii = 0; ii < 8; ++ii)
                sS[(qi * 16 + ii + 8 * hf) * 128 + wave * 16 + l16] = sacc[qi][ii];
        __syncthreads();

        // ---- online softmax stats (each wave computes identical stats) ----
#pragma unroll
        for (int qi = 0; qi < 2; ++qi) {
            const int row = qi * 16 + l16;
            const float* srow = sS + row * 128 + hf * 64;
            float mx = -1e30f;
#pragma unroll 8
            for (int c = 0; c < 64; ++c) mx = fmaxf(mx, srow[c]);
            mx = fmaxf(mx, __shfl_xor(mx, 16, 32));   // partner lane has same row
            const float mNew  = fmaxf(m[qi], mx);
            const float alpha = __expf(m[qi] - mNew);
            float s = 0.f;
            bf16_t* prow = pB + row * 128 + hf * 64;
#pragma unroll 8
            for (int c = 0; c < 64; ++c) {
                float v = __expf(srow[c] - mNew);
                s += v;
                prow[c] = (bf16_t)v;   // waves write identical values
            }
            s += __shfl_xor(s, 16, 32);
            l[qi] = l[qi] * alpha + s;
            m[qi] = mNew;
            if (wave == 0 && hf == 0) aB[row] = alpha;
        }
        __syncthreads();

        // ---- rescale accumulators, then O += P @ V (V fragment reused for both qi) ----
        float f[2][8];
#pragma unroll
        for (int qi = 0; qi < 2; ++qi)
#pragma unroll
            for (int ii = 0; ii < 8; ++ii) f[qi][ii] = aB[qi * 16 + ii + 8 * hf];
#pragma unroll
        for (int qi = 0; qi < 2; ++qi)
#pragma unroll
            for (int t = 0; t < 8; ++t)
#pragma unroll
                for (int ii = 0; ii < 8; ++ii) acc[qi][t][ii] *= f[qi][ii];

#pragma unroll
        for (int kk = 0; kk < 128; kk += 32) {
            ABFrag pf[2];
#pragma unroll
            for (int qi = 0; qi < 2; ++qi) {
                const bf16_t* pp = pB + (qi * 16 + l16) * 128 + kk + hf * 8;
                pf[qi].q[0] = *(const uint4*)pp;
                pf[qi].q[1] = *(const uint4*)(pp + 16);
            }
#pragma unroll
            for (int t = 0; t < 8; ++t) {
                const int e = wave * 128 + t * 16 + l16;
                const bf16_t* vp = Vrow + ((size_t)e << 11) + j0 + kk + hf * 8;
                ABFrag vf;
                vf.q[0] = *(const uint4*)vp;
                vf.q[1] = *(const uint4*)(vp + 16);
#pragma unroll
                for (int qi = 0; qi < 2; ++qi)
                    acc[qi][t] = wmma_bf16(pf[qi].v, vf.v, acc[qi][t]);
            }
        }
        __syncthreads();   // protect sS/pB/aB before next key block
    }

    if (wave == 0 && hf == 0) { lB[l16] = l[0]; lB[16 + l16] = l[1]; }
    __syncthreads();
    float inv[2][8];
#pragma unroll
    for (int qi = 0; qi < 2; ++qi)
#pragma unroll
        for (int ii = 0; ii < 8; ++ii) inv[qi][ii] = 1.0f / lB[qi * 16 + ii + 8 * hf];
#pragma unroll
    for (int qi = 0; qi < 2; ++qi)
#pragma unroll
        for (int t = 0; t < 8; ++t) {
            const int e = wave * 128 + t * 16 + l16;
#pragma unroll
            for (int ii = 0; ii < 8; ++ii) {
                const int r = qBase + qi * 16 + ii + 8 * hf;
                out[(((size_t)(b * SEQ + r)) << 10) + e] = acc[qi][t][ii] * inv[qi][ii];
            }
        }
}

// ---------------- launch ----------------
extern "C" void kernel_launch(void* const* d_in, const int* in_sizes, int n_in,
                              void* d_out, int out_size, void* d_ws, size_t ws_size,
                              hipStream_t stream) {
    (void)in_sizes; (void)n_in; (void)out_size; (void)ws_size;
    const float* X  = (const float*)d_in[0];
    const float* Wq = (const float*)d_in[1];
    const float* bq = (const float*)d_in[2];
    const float* Wk = (const float*)d_in[3];
    const float* bk = (const float*)d_in[4];
    const float* Wv = (const float*)d_in[5];
    const float* bv = (const float*)d_in[6];
    float* out = (float*)d_out;

    char* ws = (char*)d_ws;
    const size_t X_ELEMS = (size_t)BATCH * SEQ * DIM;       // 8M
    bf16_t* Xb = (bf16_t*)(ws);                             // 16 MB
    bf16_t* Wt = (bf16_t*)(ws + 16u * 1024 * 1024);         // 3 x 2 MB
    bf16_t* Qb = (bf16_t*)(ws + 22u * 1024 * 1024);         // 16 MB
    bf16_t* Kb = Qb + X_ELEMS;                              // 16 MB
    bf16_t* Vt = Kb + X_ELEMS;                              // 16 MB (transposed [b][e][n])

    cvt_x_kernel<<<(int)(X_ELEMS / (256 * 4)), 256, 0, stream>>>(X, Xb);
    cvt_w_kernel<<<dim3(DIM, 3), 256, 0, stream>>>(Wq, Wk, Wv, Wt);
    qkv_gemm_kernel<<<dim3((BATCH * SEQ) / 128, DIM / 256, 3), 256, 0, stream>>>(
        Xb, Wt, bq, bk, bv, Qb, Kb, Vt);
    attn_kernel<<<dim3(SEQ / 32, BATCH), 256, 0, stream>>>(Qb, Kb, Vt, out);
}
